// GATWithEdgeFeatures_81819126989063
// MI455X (gfx1250) — matrix operand order repaired and checked by reading, not compiled, not run on hardware
//
#include <hip/hip_runtime.h>
#include <hip/hip_bf16.h>

typedef __attribute__((ext_vector_type(16))) _Float16 v16h;
typedef __attribute__((ext_vector_type(8)))  float    v8f;

#define HEADS 4
#define CH 128   // HEADS*OUT_CH == IN_CH == 128

// ---------------------------------------------------------------------------
// Kernel 1: xh = x @ lin_w^T  via v_wmma_f32_16x16x32_f16
// Block = 256 threads (8 waves). Block computes a 16-row stripe of xh;
// wave w computes output columns [16w, 16w+16). K=128 -> 4 WMMA per wave.
// ---------------------------------------------------------------------------
__global__ __launch_bounds__(256)
void k_node_transform(const float* __restrict__ x, const float* __restrict__ lin_w,
                      float* __restrict__ xh, int n) {
  __shared__ float sx[16 * CH];
  const int t    = threadIdx.x;
  const int row0 = blockIdx.x * 16;

  // Stage 16x128 f32 tile of x into LDS, coalesced.
  if (row0 + 16 <= n) {
    const float4* xg = (const float4*)(x + (size_t)row0 * CH);
    float4* s4 = (float4*)sx;
    s4[t]       = xg[t];
    s4[t + 256] = xg[t + 256];
  } else {
    for (int i = t; i < 16 * CH; i += 256) {
      int r = row0 + (i >> 7);
      sx[i] = (r < n) ? x[(size_t)r * CH + (i & 127)] : 0.0f;
    }
  }
  __syncthreads();

  const int wave = t >> 5;
  const int lane = t & 31;
  const int hi   = lane >> 4;        // lane half: 0 = lanes 0-15, 1 = lanes 16-31
  const int l16  = lane & 15;
  const int ncol = wave * 16 + l16;  // output column 0..127

  v8f c = {};
  #pragma unroll
  for (int kb = 0; kb < 4; ++kb) {
    const int k0 = kb * 32;
    // A fragment (16x32 f16): row=l16; K = {k0+8*hi+v} for a[v], {k0+16+8*hi+v} for a[v+8]
    v16h a;
    const float* ar = sx + l16 * CH + k0 + hi * 8;
    #pragma unroll
    for (int v = 0; v < 8; ++v) {
      a[v]     = (_Float16)ar[v];
      a[v + 8] = (_Float16)ar[v + 16];
    }
    // B fragment (32x16 f16): B[k, ncol] = lin_w[ncol, k]; same K order as A:
    // b[v] K={k0+8*hi+v}, b[v+8] K={k0+16+8*hi+v}
    v16h b;
    const float* br = lin_w + (size_t)ncol * CH + k0;
    #pragma unroll
    for (int v = 0; v < 16; ++v) b[v] = (_Float16)br[(v & 7) + ((v >> 3) << 4) + hi * 8];

    c = __builtin_amdgcn_wmma_f32_16x16x32_f16(false, a, false, b, (short)0, c,
                                               false, false);
  }
  // D layout: VGPR r holds (M = 8*hi + r, N = l16)
  #pragma unroll
  for (int r = 0; r < 8; ++r) {
    int gr = row0 + hi * 8 + r;
    if (gr < n) xh[(size_t)gr * CH + ncol] = c[r];
  }
}

// ---------------------------------------------------------------------------
// Kernel 2: per-node attention dots  ai[n,h]=<xh[n,h,:],att_i[h,:]>, same for aj
// One wave per node; lane = channel within head; shuffle-reduce.
// ---------------------------------------------------------------------------
__global__ __launch_bounds__(256)
void k_node_dots(const float* __restrict__ xh, const float* __restrict__ att,
                 float* __restrict__ ai, float* __restrict__ aj, int n) {
  int gw   = (blockIdx.x * 256 + threadIdx.x) >> 5;
  int lane = threadIdx.x & 31;
  if (gw >= n) return;
  const float* xr = xh + (size_t)gw * CH;
  float si[HEADS], sj[HEADS];
  #pragma unroll
  for (int h = 0; h < HEADS; ++h) {
    float v = xr[h * 32 + lane];
    si[h] = v * att[h * 96 + lane];        // att_i
    sj[h] = v * att[h * 96 + 32 + lane];   // att_j
  }
  #pragma unroll
  for (int m = 16; m >= 1; m >>= 1) {
    #pragma unroll
    for (int h = 0; h < HEADS; ++h) {
      si[h] += __shfl_xor(si[h], m, 32);
      sj[h] += __shfl_xor(sj[h], m, 32);
    }
  }
  if (lane == 0) {
    #pragma unroll
    for (int h = 0; h < HEADS; ++h) {
      ai[(size_t)gw * 4 + h] = si[h];
      aj[(size_t)gw * 4 + h] = sj[h];
    }
  }
}

// ae[h] = sum_c lin_edge_w[h*32+c] * att_e[h,c]   (single wave)
__global__ void k_edge_scale(const float* __restrict__ lin_edge_w,
                             const float* __restrict__ att, float* __restrict__ ae) {
  int lane = threadIdx.x & 31;
  #pragma unroll
  for (int h = 0; h < HEADS; ++h) {
    float s = lin_edge_w[h * 32 + lane] * att[h * 96 + 64 + lane];
    #pragma unroll
    for (int m = 16; m >= 1; m >>= 1) s += __shfl_xor(s, m, 32);
    if (lane == 0) ae[h] = s;
  }
}

// ---- order-preserving f32 <-> u32 map for atomic max --------------------
__device__ __forceinline__ unsigned int f32_order(float f) {
  unsigned int u = __float_as_uint(f);
  return (u & 0x80000000u) ? ~u : (u | 0x80000000u);
}
__device__ __forceinline__ float f32_unorder(unsigned int u) {
  return __uint_as_float((u & 0x80000000u) ? (u & 0x7fffffffu) : ~u);
}

// ---------------------------------------------------------------------------
// Kernel 3: per-edge logits + leaky-relu + segment max (one thread / edge)
// ---------------------------------------------------------------------------
__global__ __launch_bounds__(256)
void k_edge_logits(const int* __restrict__ ei, const float* __restrict__ ea,
                   const float* __restrict__ ai, const float* __restrict__ aj,
                   const float* __restrict__ ae, float* __restrict__ alpha,
                   unsigned int* __restrict__ amax, int E) {
  int e = blockIdx.x * 256 + threadIdx.x;
  if (e >= E) return;
  int row = ei[e], col = ei[E + e];
  float eav = ea[e];
  float4 vi = *(const float4*)(ai + (size_t)row * 4);
  float4 vj = *(const float4*)(aj + (size_t)col * 4);
  const float* vip = (const float*)&vi;
  const float* vjp = (const float*)&vj;
  float4 al;
  float* alp = (float*)&al;
  #pragma unroll
  for (int h = 0; h < HEADS; ++h) {
    float a = vip[h] + vjp[h] + eav * ae[h];
    a = (a > 0.0f) ? a : 0.2f * a;            // leaky_relu(0.2)
    alp[h] = a;
    atomicMax(&amax[(size_t)row * 4 + h], f32_order(a));
  }
  *(float4*)(alpha + (size_t)e * 4) = al;
}

// ---------------------------------------------------------------------------
// Kernel 4: ex = exp(alpha - amax[row]); den[row] += ex  (one thread / edge)
// ---------------------------------------------------------------------------
__global__ __launch_bounds__(256)
void k_edge_exp(const int* __restrict__ ei, float* __restrict__ alpha,
                const unsigned int* __restrict__ amax, float* __restrict__ den, int E) {
  int e = blockIdx.x * 256 + threadIdx.x;
  if (e >= E) return;
  int row = ei[e];
  float4 al = *(float4*)(alpha + (size_t)e * 4);
  float* alp = (float*)&al;
  const unsigned int* mx = amax + (size_t)row * 4;
  #pragma unroll
  for (int h = 0; h < HEADS; ++h) {
    float exv = __expf(alp[h] - f32_unorder(mx[h]));
    alp[h] = exv;
    atomicAdd(&den[(size_t)row * 4 + h], exv);
  }
  *(float4*)(alpha + (size_t)e * 4) = al;
}

// ---------------------------------------------------------------------------
// Kernel 5: out[row] += (ex/den[row]) * xh[col]   (one wave / edge; lane = chan)
// xh (51 MB) and out (51 MB) both resident in 192 MB L2 -> on-die gather/scatter.
// ---------------------------------------------------------------------------
__global__ __launch_bounds__(256)
void k_scatter(const int* __restrict__ ei, const float* __restrict__ alpha,
               const float* __restrict__ den, const float* __restrict__ xh,
               float* __restrict__ out, int E) {
  int gw   = (blockIdx.x * 256 + threadIdx.x) >> 5;
  int lane = threadIdx.x & 31;
  if (gw >= E) return;
  int row = ei[gw], col = ei[E + gw];
  float4 al = *(const float4*)(alpha + (size_t)gw * 4);
  float4 dn = *(const float4*)(den + (size_t)row * 4);
  const float* alp = (const float*)&al;
  const float* dnp = (const float*)&dn;
  const float* xc = xh + (size_t)col * CH;
  float* od       = out + (size_t)row * CH;
  #pragma unroll
  for (int h = 0; h < HEADS; ++h) {
    float w = alp[h] / (dnp[h] + 1e-16f);
    atomicAdd(&od[h * 32 + lane], w * xc[h * 32 + lane]);
  }
}

// ---------------------------------------------------------------------------
extern "C" void kernel_launch(void* const* d_in, const int* in_sizes, int n_in,
                              void* d_out, int out_size, void* d_ws, size_t ws_size,
                              hipStream_t stream) {
  const float* x          = (const float*)d_in[0];
  const float* edge_attr  = (const float*)d_in[1];
  const int*   edge_index = (const int*)d_in[2];
  const float* lin_w      = (const float*)d_in[3];
  const float* lin_edge_w = (const float*)d_in[4];
  const float* att        = (const float*)d_in[5];
  float* out = (float*)d_out;

  const int N = in_sizes[0] / CH;
  const int E = in_sizes[1];

  // Workspace carve-out (256B aligned)
  char* ws = (char*)d_ws;
  size_t off = 0;
  auto carve = [&](size_t bytes) -> char* {
    char* p = ws + off;
    off += (bytes + 255) & ~(size_t)255;
    return p;
  };
  float*        xh    = (float*)carve((size_t)N * CH * 4);
  float*        ai    = (float*)carve((size_t)N * 4 * 4);
  float*        aj    = (float*)carve((size_t)N * 4 * 4);
  float*        ae    = (float*)carve(256);
  unsigned int* amax  = (unsigned int*)carve((size_t)N * 4 * 4);
  float*        den   = (float*)carve((size_t)N * 4 * 4);
  float*        alpha = (float*)carve((size_t)E * 4 * 4);

  // Zero accumulators (graph-capturable async memsets)
  hipMemsetAsync(out,  0, (size_t)N * CH * 4, stream);
  hipMemsetAsync(amax, 0, (size_t)N * 4 * 4, stream);   // 0 == ordered(-inf) floor
  hipMemsetAsync(den,  0, (size_t)N * 4 * 4, stream);

  const int gemm_blocks = (N + 15) / 16;
  k_node_transform<<<gemm_blocks, 256, 0, stream>>>(x, lin_w, xh, N);

  const int dot_blocks = ((size_t)N * 32 + 255) / 256;
  k_node_dots<<<dot_blocks, 256, 0, stream>>>(xh, att, ai, aj, N);
  k_edge_scale<<<1, 32, 0, stream>>>(lin_edge_w, att, ae);

  const int edge_blocks = (E + 255) / 256;
  k_edge_logits<<<edge_blocks, 256, 0, stream>>>(edge_index, edge_attr, ai, aj, ae,
                                                 alpha, amax, E);
  k_edge_exp<<<edge_blocks, 256, 0, stream>>>(edge_index, alpha, amax, den, E);

  const int scat_blocks = ((size_t)E * 32 + 255) / 256;
  k_scatter<<<scat_blocks, 256, 0, stream>>>(edge_index, alpha, den, xh, out, E);
}
